// APPNPNet_65180423684247
// MI455X (gfx1250) — matrix operand order repaired and checked by reading, not compiled, not run on hardware
//
#include <hip/hip_runtime.h>
#include <hip/hip_bf16.h>

// ---------------------------------------------------------------------------
// Fully fused APPNP network for MI455X (gfx1250), one workgroup per batch row.
// All GEMMs run on v_wmma_f32_16x16x32_bf16 (f32 accumulate).
// APPNP is linear: x_K = Q x0 with Q = a*sum_{k<10}((1-a)S)^k + ((1-a)S)^10,
// a 32x32 matrix built once per block. Hop phase #1 = one WMMA pass,
// hop phase #2 is commuted through the scalar readout (Q h)Wl = Q (h Wl).
// Each wave owns 4 exclusive N-tiles x both M-tiles so every weight tile is
// fetched exactly once per block (halves L2 weight streaming).
// ---------------------------------------------------------------------------

typedef __attribute__((ext_vector_type(16))) __bf16 v16bf;
typedef __attribute__((ext_vector_type(8)))  __bf16 v8bf;
typedef __attribute__((ext_vector_type(8)))  float  v8f;

#define N_NODES 30
#define NP      32          // padded node count (2 x 16 tiles)
#define IC      256
#define FDIM    512
#define CDIM    4
#define HFREQ   5
#define KHOPS   10
#define ALPHA   0.1f

__device__ __forceinline__ float bf2f(unsigned short h) {
  union { unsigned u; float f; } v; v.u = ((unsigned)h) << 16;
  return v.f;
}

// Pack two floats into two bf16 (low = a, high = b) with RNE: one VOP3.
__device__ __forceinline__ unsigned pack2bf(float a, float b) {
  unsigned r;
  asm("v_cvt_pk_bf16_f32 %0, %1, %2" : "=v"(r) : "v"(a), "v"(b));
  return r;
}

union FragB16 { v16bf v; v8bf h[2]; };

// A-matrix 16x32 bf16 fragment (ISA 7.12.2 layout):
// lane L: row m = mtile*16 + (L&15); half = L>>4.
// elements 0..7  = K in [half*8 .. half*8+7]        (chunk 0)
// elements 8..15 = K in [16+half*8 .. 16+half*8+7]  (chunk 1)
__device__ __forceinline__ v16bf load_a_frag(const unsigned short* base,
                                             int mrow, int kbase, int stride,
                                             int half) {
  FragB16 f;
  const unsigned short* p = base + mrow * stride + kbase + half * 8;
  f.h[0] = ((const v8bf*)p)[0];
  f.h[1] = ((const v8bf*)(p + 16))[0];
  return f.v;
}

// B-matrix 32x16 bf16 fragment: lane L holds column N=L&15; p points at row
// K=half*8 of that column; elements 0..7 = K half*8.., 8..15 = K 16+half*8..
__device__ __forceinline__ v16bf load_b_frag16(const unsigned short* p,
                                               int stride) {
  union { v16bf v; unsigned short u[16]; } f;
  #pragma unroll
  for (int s = 0; s < 8; ++s) {
    f.u[s]     = p[s * stride];
    f.u[8 + s] = p[(16 + s) * stride];
  }
  return f.v;
}

extern "C" __global__ void f32_to_bf16_kernel(const float* __restrict__ src,
                                              unsigned* __restrict__ dst,
                                              int npairs) {
  int i = blockIdx.x * blockDim.x + threadIdx.x;
  if (i < npairs) dst[i] = pack2bf(src[2 * i], src[2 * i + 1]);
}

// One layer: dstH = relu(src (32xKDIM) @ W (KDIMx512) + bias), bf16 into LDS.
// Wave covers N-tiles [ntile0, ntile0+4) x both M-tiles; each W tile loaded
// once and reused for two WMMAs.
template <int KDIM>
__device__ __forceinline__ void layer_gemm(const unsigned short* src,
                                           const unsigned short* __restrict__ Wbf,
                                           const float* __restrict__ bias,
                                           unsigned short* dstH,
                                           int ntile0, int lane,
                                           int half, int l15) {
  v8f acc0[4], acc1[4];
  #pragma unroll
  for (int j = 0; j < 4; ++j) { acc0[j] = (v8f){}; acc1[j] = (v8f){}; }
  #pragma unroll
  for (int kk = 0; kk < KDIM / 32; ++kk) {
    v16bf a0 = load_a_frag(src, l15,      kk * 32, KDIM, half);  // rows 0..15
    v16bf a1 = load_a_frag(src, 16 + l15, kk * 32, KDIM, half);  // rows 16..31
    const unsigned short* bp = Wbf + (size_t)(kk * 32 + half * 8) * FDIM + l15;
    #pragma unroll
    for (int j = 0; j < 4; ++j) {
      v16bf bb = load_b_frag16(bp + (ntile0 + j) * 16, FDIM);
      acc0[j] = __builtin_amdgcn_wmma_f32_16x16x32_bf16(false, a0, false, bb,
                                                        (short)0, acc0[j],
                                                        false, false);
      acc1[j] = __builtin_amdgcn_wmma_f32_16x16x32_bf16(false, a1, false, bb,
                                                        (short)0, acc1[j],
                                                        false, false);
    }
  }
  const int m0 = half * 8;
  #pragma unroll
  for (int j = 0; j < 4; ++j) {
    int n = (ntile0 + j) * 16 + l15;
    float bs = bias[n];
    #pragma unroll
    for (int r = 0; r < 8; r += 2) {
      unsigned p0 = pack2bf(fmaxf(acc0[j][r] + bs, 0.0f),
                            fmaxf(acc0[j][r + 1] + bs, 0.0f));
      dstH[(m0 + r) * FDIM + n]          = (unsigned short)(p0 & 0xffffu);
      dstH[(m0 + r + 1) * FDIM + n]      = (unsigned short)(p0 >> 16);
      unsigned p1 = pack2bf(fmaxf(acc1[j][r] + bs, 0.0f),
                            fmaxf(acc1[j][r + 1] + bs, 0.0f));
      dstH[(16 + m0 + r) * FDIM + n]     = (unsigned short)(p1 & 0xffffu);
      dstH[(16 + m0 + r + 1) * FDIM + n] = (unsigned short)(p1 >> 16);
    }
  }
}

// nxt = Q @ cur over the 512 feature columns (one WMMA pass, both M-tiles).
__device__ __forceinline__ void apply_q(const unsigned short* cur,
                                        unsigned short* nxt,
                                        v16bf qf0, v16bf qf1,
                                        int ntile0, int lane,
                                        int half, int l15) {
  const int m0 = half * 8;
  const v8f z = {};
  #pragma unroll
  for (int j = 0; j < 4; ++j) {
    int ntile = ntile0 + j;
    v16bf bb = load_b_frag16(cur + half * 8 * FDIM + ntile * 16 + l15, FDIM);
    v8f acc0 = __builtin_amdgcn_wmma_f32_16x16x32_bf16(false, qf0, false, bb,
                                                       (short)0, z, false, false);
    v8f acc1 = __builtin_amdgcn_wmma_f32_16x16x32_bf16(false, qf1, false, bb,
                                                       (short)0, z, false, false);
    int n = ntile * 16 + l15;
    #pragma unroll
    for (int r = 0; r < 8; r += 2) {
      unsigned p0 = pack2bf(acc0[r], acc0[r + 1]);
      nxt[(m0 + r) * FDIM + n]          = (unsigned short)(p0 & 0xffffu);
      nxt[(m0 + r + 1) * FDIM + n]      = (unsigned short)(p0 >> 16);
      unsigned p1 = pack2bf(acc1[r], acc1[r + 1]);
      nxt[(16 + m0 + r) * FDIM + n]     = (unsigned short)(p1 & 0xffffu);
      nxt[(16 + m0 + r + 1) * FDIM + n] = (unsigned short)(p1 >> 16);
    }
  }
}

extern "C" __global__ __launch_bounds__(256)
void appnp_fused_kernel(const float* __restrict__ real,
                        const float* __restrict__ graph,
                        const unsigned short* __restrict__ W1bf,
                        const float* __restrict__ b1,
                        const unsigned short* __restrict__ W2bf,
                        const float* __restrict__ b2,
                        const float* __restrict__ Wl,
                        const float* __restrict__ bl,
                        const float* __restrict__ Wc,
                        const float* __restrict__ bc,
                        float* __restrict__ out) {
  __shared__ __align__(16) float          sMt[NP * NP];    // 4 KB: Mt, then Q (f32)
  __shared__ __align__(16) float          sdinv[NP];
  __shared__ __align__(16) unsigned short sSbf[NP * NP];   // 2 KB: S' = (1-a)S bf16
  __shared__ __align__(16) unsigned short sPa[NP * NP];    // 2 KB: S' power ping
  __shared__ __align__(16) unsigned short sPb[NP * NP];    // 2 KB: S' power pong
  __shared__ __align__(16) unsigned short sHa[NP * FDIM];  // 32 KB
  __shared__ __align__(16) unsigned short sHb[NP * FDIM];  // 32 KB
  __shared__ __align__(16) float          sred[256];
  __shared__ __align__(16) float          sxvec[NP];

  unsigned short* sXin = sHb;   // 32x256 bf16 input tile aliases sHb
  float*          sQ32 = sMt;   // f32 Q reuses Mt storage

  const int b    = blockIdx.x;
  const int tid  = threadIdx.x;
  const int lane = tid & 31;
  const int wv   = tid >> 5;
  const int half = lane >> 4;
  const int l15  = lane & 15;
  const int ntile0 = wv * 4;    // 4 exclusive N-tiles per wave, both M-tiles

  // Warm L2 for the (block-shared) weight matrices.
  __builtin_prefetch(W1bf + tid * 128, 0, 1);
  __builtin_prefetch(W2bf + tid * 128, 0, 1);

  // ---------------- Phase 0: S' = (1-a) * gcn_norm operator ----------------
  for (int i = tid; i < NP * NP; i += 256) sMt[i] = 0.0f;
  __syncthreads();
  for (int idx = tid; idx < N_NODES * N_NODES; idx += 256) {
    int i = idx / N_NODES, j = idx % N_NODES;
    const float* gp = graph + (((size_t)b * HFREQ) * N_NODES + i) * N_NODES + j;
    float s = 0.0f;
    #pragma unroll
    for (int h = 0; h < HFREQ; ++h) s += gp[h * N_NODES * N_NODES];
    float m = (s != 0.0f) ? 1.0f : 0.0f;   // (mean != 0) <=> (sum != 0)
    if (i == j) m += 1.0f;                 // self loop
    sMt[i * NP + j] = m;
  }
  __syncthreads();
  if (tid < NP) {
    float d = 0.0f;
    if (tid < N_NODES)
      for (int i = 0; i < N_NODES; ++i) d += sMt[i * NP + tid];  // column sums
    sdinv[tid] = (tid < N_NODES) ? rsqrtf(d) : 0.0f;             // deg >= 1
  }
  __syncthreads();
  for (int idx = tid; idx < NP * NP / 2; idx += 256) {
    int i = idx >> 4, j = (idx & 15) * 2;
    float di = (1.0f - ALPHA) * sdinv[i];
    float s0 = di * sMt[j * NP + i] * sdinv[j];        // S' = (1-a) Dinv Mt^T Dinv
    float s1 = di * sMt[(j + 1) * NP + i] * sdinv[j + 1];
    ((unsigned*)sSbf)[idx] = pack2bf(s0, s1);          // padded rows/cols = 0
  }
  __syncthreads();                         // sMt free from here; becomes Q acc

  // ---------------- Phase 1: Q accumulator init + stage real[b] -----------
  for (int idx = tid; idx < NP * NP; idx += 256)
    sQ32[idx] = ((idx >> 5) == (idx & 31)) ? ALPHA : 0.0f;   // a * I  (k = 0)
  for (int idx = tid; idx < NP * IC / 2; idx += 256) {
    int row = idx >> 7, col2 = (idx & 127) * 2;
    float v0 = 0.0f, v1 = 0.0f;
    if (row < N_NODES) {
      const float* rp = real + ((size_t)b * N_NODES + row) * IC + col2;
      v0 = rp[0]; v1 = rp[1];
    }
    ((unsigned*)sXin)[idx] = pack2bf(v0, v1);
  }
  __syncthreads();

  // ---------------- Phase 2: build Q ----------------
  // Pcur = P_k = S'^k; per step: Q += a*P_k (all waves), P_{k+1} = P_k @ S'
  // (waves 0..3, one 16x16 tile each). Ping-pong selected by an int (no
  // LDS-pointer arrays: addrspacecast aggregates break the linker).
  const unsigned short* Pcur = sSbf;                    // P_1
  int sel = 0;
  {
    const v8f z = {};
    for (int k = 1; k <= KHOPS - 1; ++k) {              // k = 1..9
      for (int idx = tid; idx < NP * NP; idx += 256)
        sQ32[idx] += ALPHA * bf2f(Pcur[idx]);
      unsigned short* Pdst = sel ? sPb : sPa;
      if (wv < 4) {
        int mt = wv & 1, nt = wv >> 1;
        v16bf a  = load_a_frag(Pcur, mt * 16 + l15, 0, NP, half);
        v16bf bb = load_b_frag16(sSbf + half * 8 * NP + nt * 16 + l15, NP);
        v8f acc = __builtin_amdgcn_wmma_f32_16x16x32_bf16(false, a, false, bb,
                                                          (short)0, z,
                                                          false, false);
        int n = nt * 16 + l15, m0 = mt * 16 + half * 8;
        #pragma unroll
        for (int r = 0; r < 8; r += 2) {
          unsigned p = pack2bf(acc[r], acc[r + 1]);
          Pdst[(m0 + r) * NP + n]     = (unsigned short)(p & 0xffffu);
          Pdst[(m0 + r + 1) * NP + n] = (unsigned short)(p >> 16);
        }
      }
      __syncthreads();
      Pcur = Pdst;
      sel ^= 1;
    }
  }
  // Q = a*sum_{k=0..9} P_k + P_10 ; keep f32 copy and bf16 copy.
  // After the loop `sel` selects the buffer NOT holding Pcur (= P_10).
  unsigned short* sQbf = sel ? sPb : sPa;
  for (int idx = tid; idx < NP * NP / 2; idx += 256) {
    int i2 = idx * 2;
    float q0 = sQ32[i2]     + bf2f(Pcur[i2]);
    float q1 = sQ32[i2 + 1] + bf2f(Pcur[i2 + 1]);
    sQ32[i2] = q0; sQ32[i2 + 1] = q1;
    ((unsigned*)sQbf)[idx] = pack2bf(q0, q1);
  }
  __syncthreads();
  v16bf qf0 = load_a_frag(sQbf, l15,      0, NP, half);   // Q rows 0..15
  v16bf qf1 = load_a_frag(sQbf, 16 + l15, 0, NP, half);   // Q rows 16..31

  // ---------------- Phase 3: layer 1 (30x256 @ 256x512) -> sHa ------------
  layer_gemm<IC>(sXin, W1bf, b1, sHa, ntile0, lane, half, l15);
  __syncthreads();

  // ---------------- Phase 4: APPNP #1 as one pass: sHb = Q @ sHa ----------
  apply_q(sHa, sHb, qf0, qf1, ntile0, lane, half, l15);
  __syncthreads();

  // ---------------- Phase 5: layer 2 (30x512 @ 512x512) -> sHa ------------
  layer_gemm<FDIM>(sHb, W2bf, b2, sHa, ntile0, lane, half, l15);
  __syncthreads();

  // ---------------- Phase 6: readout (APPNP #2 commuted through Wl) -------
  // v = h2 @ Wl   (32-vector; padded rows filtered by Q's zero columns)
  if (tid < N_NODES * 8) {                 // 240 threads: (node, 64-wide seg)
    int m = tid >> 3, seg = tid & 7;
    const unsigned short* hp = sHa + m * FDIM + seg * 64;
    const float* wp = Wl + seg * 64;
    float s = 0.0f;
    #pragma unroll 8
    for (int f = 0; f < 64; ++f) s += bf2f(hp[f]) * wp[f];
    sred[tid] = s;
  }
  __syncthreads();
  if (tid < NP) {
    float v = 0.0f;
    if (tid < N_NODES)
      for (int k = 0; k < 8; ++k) v += sred[tid * 8 + k];
    sxvec[tid] = v;
  }
  __syncthreads();
  if (tid < NP) {                          // y = Q v ; x = relu(y + bl)
    float y = 0.0f;
    for (int j = 0; j < N_NODES; ++j) y += sQ32[tid * NP + j] * sxvec[j];
    sred[tid] = fmaxf(y + bl[0], 0.0f);
  }
  __syncthreads();
  if (tid < CDIM) {
    float s = bc[tid];
    for (int m = 0; m < N_NODES; ++m) s += sred[m] * Wc[tid * N_NODES + m];
    out[(size_t)b * CDIM + tid] = s;
  }
}

extern "C" void kernel_launch(void* const* d_in, const int* in_sizes, int n_in,
                              void* d_out, int out_size, void* d_ws, size_t ws_size,
                              hipStream_t stream) {
  const float* real  = (const float*)d_in[0];
  /* d_in[1] = imag: unused by the reference */
  const float* graph = (const float*)d_in[2];
  const float* W1 = (const float*)d_in[3];
  const float* b1 = (const float*)d_in[4];
  const float* W2 = (const float*)d_in[5];
  const float* b2 = (const float*)d_in[6];
  const float* Wl = (const float*)d_in[7];
  const float* bl = (const float*)d_in[8];
  const float* Wc = (const float*)d_in[9];
  const float* bc = (const float*)d_in[10];

  unsigned short* W1bf = (unsigned short*)d_ws;
  unsigned short* W2bf = (unsigned short*)((char*)d_ws + (size_t)IC * FDIM * 2);

  const int p1 = IC * FDIM / 2;       // 65536 bf16 pairs
  const int p2 = FDIM * FDIM / 2;     // 131072 bf16 pairs
  f32_to_bf16_kernel<<<(p1 + 255) / 256, 256, 0, stream>>>(W1, (unsigned*)W1bf, p1);
  f32_to_bf16_kernel<<<(p2 + 255) / 256, 256, 0, stream>>>(W2, (unsigned*)W2bf, p2);

  const int B = in_sizes[0] / (N_NODES * IC);   // 4096
  appnp_fused_kernel<<<B, 256, 0, stream>>>(real, graph, W1bf, b1, W2bf, b2,
                                            Wl, bl, Wc, bc, (float*)d_out);
}
